// GGTP_Decoder_38474317038225
// MI455X (gfx1250) — matrix operand order repaired and checked by reading, not compiled, not run on hardware
//
#include <hip/hip_runtime.h>
#include <hip/hip_bf16.h>
#include <stdint.h>

typedef _Float16 half_t;
typedef __attribute__((ext_vector_type(16))) _Float16 v16h;
typedef __attribute__((ext_vector_type(8)))  _Float16 v8h;
typedef __attribute__((ext_vector_type(8)))  float    v8f;
typedef __attribute__((ext_vector_type(4)))  unsigned int u32x4;
typedef __attribute__((ext_vector_type(8)))  int      i32x8;
typedef __attribute__((ext_vector_type(4)))  int      i32x4;

#ifndef __has_builtin
#define __has_builtin(x) 0
#endif
#if __has_builtin(__builtin_amdgcn_tensor_load_to_lds)
#define HAVE_TDM 1
#else
#define HAVE_TDM 0
#endif

#define B_     64
#define M_     30
#define N_     10
#define T_     80
#define TD_    3
#define AGENT_ 256
#define NA_    11
#define NP_    4
#define LAT_   32
#define FUT_   240

// LDS chunk-row stride: 16 chunks * 32B + 16B TDM pad = 528 bytes
#define BROW_  528

// ---------------------------------------------------------------------------
// Weight pack: f32 (K x N, row-major, ld=srcLd) slice -> f16 transposed (N x K)
// ---------------------------------------------------------------------------
__global__ __launch_bounds__(256)
void pack_wt(const float* __restrict__ src, int rowOfs, int Ksub, int Ncols,
             int srcLd, half_t* __restrict__ dst, int Kdst, int kOfs, int nOfs)
{
    int id = blockIdx.x * 256 + threadIdx.x;
    if (id >= Ksub * Ncols) return;
    int k = id % Ksub;
    int n = id / Ksub;
    dst[(size_t)(n + nOfs) * Kdst + kOfs + k] =
        (half_t)src[(size_t)(rowOfs + k) * srcLd + n];
}

// ---------------------------------------------------------------------------
// Ego layer1 + mean over T:  hbar[b*M+m, c] = mean_t relu(x[t,:6]·W1[:,c]+b1[c])
// ---------------------------------------------------------------------------
__global__ __launch_bounds__(128)
void ego1_kernel(const float* __restrict__ x, const float* __restrict__ W1,
                 const float* __restrict__ b1, half_t* __restrict__ hbar)
{
    __shared__ float xs[T_ * 6];
    const int bm = blockIdx.x;
    const float* xr = x + (size_t)bm * T_ * 6;
    for (int i = threadIdx.x; i < T_ * 6; i += 128) xs[i] = xr[i];
    __syncthreads();
    const int c = threadIdx.x;
    float w[6];
#pragma unroll
    for (int k = 0; k < 6; ++k) w[k] = W1[k * 128 + c];
    const float bb = b1[c];
    float s = 0.0f;
    for (int t = 0; t < T_; ++t) {
        float a = bb;
#pragma unroll
        for (int k = 0; k < 6; ++k) a += xs[t * 6 + k] * w[k];
        s += fmaxf(a, 0.0f);
    }
    hbar[(size_t)bm * 128 + c] = (half_t)(s * (1.0f / T_));
}

// ---------------------------------------------------------------------------
// Build per-(b,n) feature rows (f16):
//   Fpost (640 x 1536) = [target 256 | payoff 1024 | gt 240 | pad 16]
//   Fgen  (640 x 1280) = [target 256 | payoff 1024]
// ---------------------------------------------------------------------------
__global__ __launch_bounds__(256)
void build_features(const float* __restrict__ enc, const float* __restrict__ gtf,
                    const int* __restrict__ nidx, const unsigned char* __restrict__ nmask,
                    const int* __restrict__ tsPtr,
                    half_t* __restrict__ Fpost, half_t* __restrict__ Fgen)
{
    const int bn = blockIdx.x;
    const int b = bn / N_, n = bn % N_;
    half_t* fp = Fpost + (size_t)bn * 1536;
    half_t* fg = Fgen + (size_t)bn * 1280;
    const float* tgt = enc + ((size_t)b * NA_ + 1 + n) * AGENT_;
    for (int c = threadIdx.x; c < AGENT_; c += 256) {
        half_t v = (half_t)tgt[c];
        fp[c] = v; fg[c] = v;
    }
    for (int i = threadIdx.x; i < NP_ * AGENT_; i += 256) {
        int p = i / AGENT_, c = i % AGENT_;
        int a = nidx[(size_t)bn * NP_ + p];
        float mv = nmask[(size_t)bn * NP_ + p] ? 1.0f : 0.0f;
        half_t v = (half_t)(enc[((size_t)b * NA_ + a) * AGENT_ + c] * mv);
        fp[AGENT_ + i] = v; fg[AGENT_ + i] = v;
    }
    int lim = tsPtr[0] * TD_;
    if (lim > FUT_) lim = FUT_;
    for (int i = threadIdx.x; i < 256; i += 256) {
        float v = (i < lim) ? gtf[(size_t)bn * FUT_ + i] : 0.0f;
        fp[1280 + i] = (half_t)v;
    }
}

// ---------------------------------------------------------------------------
// hp1[r, j] = relu(Pbn[(b,n), j] + Pbm[(b,m), j] + bias[j])
// ---------------------------------------------------------------------------
__global__ __launch_bounds__(256)
void assemble_hp1(const float* __restrict__ Pbn, const float* __restrict__ Pbm,
                  const float* __restrict__ bias, half_t* __restrict__ out)
{
    int id = blockIdx.x * 256 + threadIdx.x;
    if (id >= (B_ * M_ * N_) * 512) return;
    int row = id / 512, col = id % 512;
    int b = row / (M_ * N_);
    int m = (row / N_) % M_;
    int n = row % N_;
    float v = Pbn[((size_t)(b * N_ + n)) * 512 + col] +
              Pbm[((size_t)(b * M_ + m)) * 512 + col] + bias[col];
    out[id] = (half_t)fmaxf(v, 0.0f);
}

// ---------------------------------------------------------------------------
// z[r, c] = (mu + b_mu) + eps * exp(0.5*(lv + b_lv))
// ---------------------------------------------------------------------------
__global__ __launch_bounds__(256)
void z_kernel(const float* __restrict__ mulv, const float* __restrict__ bmu,
              const float* __restrict__ blv, const float* __restrict__ eps,
              half_t* __restrict__ z16)
{
    int id = blockIdx.x * 256 + threadIdx.x;
    if (id >= (B_ * M_ * N_) * LAT_) return;
    int r = id / LAT_, c = id % LAT_;
    float mu = mulv[(size_t)r * 64 + c] + bmu[c];
    float lv = mulv[(size_t)r * 64 + 32 + c] + blv[c];
    z16[id] = (half_t)(mu + eps[id] * __expf(0.5f * lv));
}

// ---------------------------------------------------------------------------
// WMMA GEMM, TDM-staged weights:
//   C(Mr x Nc) = A(Mr x K, f16 row-major) * Bt^T,  Bt is (Nc x K) f16.
// Block = 4 waves; the 16-column weight strip for this block's N-tile is
// DMA'd into LDS once by the Tensor Data Mover in chunk-interleaved layout:
//   LDS[(c*16 + col)*32B + c*16B pad],  chunk c holds K=[16c..16c+15] of col.
// Each wave computes a 64x16 tile: A frags from global (b128 clauses),
// B frags from LDS (ds_load_b128), f32 accumulate in v_wmma.
// Requires: Nc%16==0, K%32==0, Mr%64==0.
// ---------------------------------------------------------------------------
__global__ __launch_bounds__(128)
void gemm16(const half_t* __restrict__ A, const half_t* __restrict__ Bt,
            int Mr, int Nc, int K,
            const float* __restrict__ bias, int doRelu,
            const float* __restrict__ exBN, const float* __restrict__ exBM,
            int dimM, int dimN,
            float* __restrict__ outF, half_t* __restrict__ outH,
            const int* __restrict__ tsPtr)
{
    extern __shared__ char smem[];
    const int lane = threadIdx.x & 31;
    const int wave = threadIdx.x >> 5;
    const int col0 = blockIdx.x * 16;
    const int row0 = (blockIdx.y * 4 + wave) * 64;
    const int C = K >> 4;                 // 16-half chunks per column

#if HAVE_TDM
    if (wave == 0) {
        const unsigned long long ga =
            (unsigned long long)(uintptr_t)(Bt + (size_t)col0 * K);
        u32x4 g0; i32x8 g1; i32x4 g2, g3;
        g0[0] = 1u;                                   // count=1, user D#
        g0[1] = 0u;                                   // lds_addr = 0
        g0[2] = (unsigned)(ga & 0xFFFFFFFFu);
        g0[3] = ((unsigned)(ga >> 32) & 0x01FFFFFFu) | 0x80000000u; // type=2
        // data_size=4B(2), pad_enable, pad_interval=128dw(6), pad_amount=4dw(3)
        g1[0] = (2 << 16) | (1 << 20) | (6 << 22) | (3 << 25);
        g1[1] = (8 << 16);                            // tensor_dim0 = 8 dw
        g1[2] = (16 << 16);                           // tensor_dim1 = 16
        g1[3] = (8 << 16);                            // tile_dim0 = 8 dw
        g1[4] = 16 | (C << 16);                       // tile_dim1=16, tile_dim2=C
        g1[5] = (K >> 1);                             // dim0 stride = K/2 dw
        g1[6] = (8 << 16);                            // dim1 stride = 8 dw
        g1[7] = 0;
        g2[0] = C; g2[1] = 0; g2[2] = 0; g2[3] = 0;   // tensor_dim2 = C
        g3[0] = 0; g3[1] = 0; g3[2] = 0; g3[3] = 0;
#if __clang_major__ >= 23
        i32x8 g4 = {0, 0, 0, 0, 0, 0, 0, 0};
        __builtin_amdgcn_tensor_load_to_lds(g0, g1, g2, g3, g4, 0);
#else
        __builtin_amdgcn_tensor_load_to_lds(g0, g1, g2, g3, 0);
#endif
        __builtin_amdgcn_s_wait_tensorcnt(0);
    }
#else
    // Fallback: cooperative copy into the same chunk-interleaved layout.
    for (int i = threadIdx.x; i < C * 32; i += 128) {
        int c  = i >> 5;
        int cl = (i >> 1) & 15;
        int h8 = i & 1;
        v8h v = *(const v8h*)(Bt + (size_t)(col0 + cl) * K + c * 16 + h8 * 8);
        *(v8h*)(smem + (size_t)c * BROW_ + cl * 32 + h8 * 16) = v;
    }
#endif
    __syncthreads();

    if (row0 < Mr) {
        const int arow  = lane & 15;
        const int ahalf = lane >> 4;
        v8f acc[4] = {};
        const half_t* abase = A + (size_t)(row0 + arow) * K + ahalf * 16;
        const char* bbase = smem + (size_t)arow * 32;     // col = lane&15

        for (int kk = 0; kk < K; kk += 32) {
            const char* bp = bbase + (size_t)((kk >> 4) + ahalf) * BROW_;
            v8h blo = *(const v8h*)(bp);
            v8h bhi = *(const v8h*)(bp + 16);
            v16h bf = __builtin_shufflevector(blo, bhi,
                0, 1, 2, 3, 4, 5, 6, 7, 8, 9, 10, 11, 12, 13, 14, 15);
#pragma unroll
            for (int t = 0; t < 4; ++t) {
                const half_t* ap = abase + (size_t)t * 16 * K + kk;
                v8h lo = *(const v8h*)(ap);
                v8h hi = *(const v8h*)(ap + 8);
                v16h af = __builtin_shufflevector(lo, hi,
                    0, 1, 2, 3, 4, 5, 6, 7, 8, 9, 10, 11, 12, 13, 14, 15);
                acc[t] = __builtin_amdgcn_wmma_f32_16x16x32_f16(
                    false, af, false, bf, (short)0, acc[t], false, false);
            }
        }

        const int col = col0 + arow;
        const float bval = bias ? bias[col] : 0.0f;
        int lim = 0;
        if (tsPtr) { lim = tsPtr[0] * TD_; if (lim > Nc) lim = Nc; }

#pragma unroll
        for (int t = 0; t < 4; ++t) {
#pragma unroll
            for (int r = 0; r < 8; ++r) {
                int row = row0 + t * 16 + r + ahalf * 8;
                float v = acc[t][r] + bval;
                if (exBN) {
                    int bI = row / (dimM * dimN);
                    int mI = (row / dimN) % dimM;
                    int nI = row % dimN;
                    v += exBN[((size_t)(bI * dimN + nI)) * Nc + col];
                    v += exBM[((size_t)(bI * dimM + mI)) * Nc + col];
                }
                if (doRelu) v = fmaxf(v, 0.0f);
                if (outH) outH[(size_t)row * Nc + col] = (half_t)v;
                if (tsPtr) {
                    if (col < lim) outF[(size_t)row * lim + col] = v;
                } else if (outF) {
                    outF[(size_t)row * Nc + col] = v;
                }
            }
        }
    }
}

// ---------------------------------------------------------------------------
extern "C" void kernel_launch(void* const* d_in, const int* in_sizes, int n_in,
                              void* d_out, int out_size, void* d_ws, size_t ws_size,
                              hipStream_t stream)
{
    const float* enc    = (const float*)d_in[0];
    const float* ego    = (const float*)d_in[1];
    const float* gtf    = (const float*)d_in[2];
    const float* eps    = (const float*)d_in[3];
    const int*   nidx   = (const int*)d_in[4];
    const unsigned char* nmask = (const unsigned char*)d_in[5];
    const int*   tsp    = (const int*)d_in[6];
    const float* W_ego1 = (const float*)d_in[7];
    const float* b_ego1 = (const float*)d_in[8];
    const float* W_ego2 = (const float*)d_in[9];
    const float* b_ego2 = (const float*)d_in[10];
    const float* W_post1= (const float*)d_in[11];
    const float* b_post1= (const float*)d_in[12];
    const float* W_post2= (const float*)d_in[13];
    const float* b_post2= (const float*)d_in[14];
    const float* W_mu   = (const float*)d_in[15];
    const float* b_mu   = (const float*)d_in[16];
    const float* W_lv   = (const float*)d_in[17];
    const float* b_lv   = (const float*)d_in[18];
    const float* W_gen1 = (const float*)d_in[19];
    const float* b_gen1 = (const float*)d_in[20];
    const float* W_gen2 = (const float*)d_in[21];
    const float* b_gen2 = (const float*)d_in[22];
    const float* W_gen3 = (const float*)d_in[23];
    const float* b_gen3 = (const float*)d_in[24];

    char* ws = (char*)d_ws;
    size_t cur = 0;
    auto alloc = [&](size_t bytes) -> void* {
        void* p = ws + cur;
        cur = (cur + bytes + 255) & ~((size_t)255);
        return p;
    };

    const int RBMN = B_ * M_ * N_;   // 19200
    const int RBN  = B_ * N_;        // 640
    const int RBM  = B_ * M_;        // 1920

    half_t* Wt_p1bn = (half_t*)alloc((size_t)512 * 1536 * 2);
    half_t* Wt_p1eg = (half_t*)alloc((size_t)512 * 256 * 2);
    half_t* Wt_p2   = (half_t*)alloc((size_t)256 * 512 * 2);
    half_t* Wt_mulv = (half_t*)alloc((size_t)64 * 256 * 2);
    half_t* Wt_g1bn = (half_t*)alloc((size_t)512 * 1280 * 2);
    half_t* Wt_g1eg = (half_t*)alloc((size_t)512 * 256 * 2);
    half_t* Wt_z    = (half_t*)alloc((size_t)512 * 32 * 2);
    half_t* Wt_g2   = (half_t*)alloc((size_t)256 * 512 * 2);
    half_t* Wt_g3   = (half_t*)alloc((size_t)240 * 256 * 2);
    half_t* Wt_eg2  = (half_t*)alloc((size_t)256 * 128 * 2);
    half_t* hbar16  = (half_t*)alloc((size_t)RBM * 128 * 2);
    half_t* Fpost   = (half_t*)alloc((size_t)RBN * 1536 * 2);
    half_t* Fgen    = (half_t*)alloc((size_t)RBN * 1280 * 2);
    half_t* ego16   = (half_t*)alloc((size_t)RBM * 256 * 2);
    half_t* hp1     = (half_t*)alloc((size_t)RBMN * 512 * 2);  // reused as hg1
    half_t* hp2     = (half_t*)alloc((size_t)RBMN * 256 * 2);  // reused as hg2
    half_t* z16     = (half_t*)alloc((size_t)RBMN * LAT_ * 2);
    float* Pbn  = (float*)alloc((size_t)RBN * 512 * 4);
    float* Gbn  = (float*)alloc((size_t)RBN * 512 * 4);
    float* Pbm  = (float*)alloc((size_t)RBM * 512 * 4);
    float* Gbm  = (float*)alloc((size_t)RBM * 512 * 4);
    float* mulv = (float*)alloc((size_t)RBMN * 64 * 4);
    (void)ws_size; (void)in_sizes; (void)n_in; (void)out_size;

    auto pack = [&](const float* src, int rowOfs, int Ksub, int Ncols, int ld,
                    half_t* dst, int Kdst, int kOfs, int nOfs) {
        int tot = Ksub * Ncols;
        pack_wt<<<(tot + 255) / 256, 256, 0, stream>>>(
            src, rowOfs, Ksub, Ncols, ld, dst, Kdst, kOfs, nOfs);
    };
    auto gemm = [&](const half_t* A, const half_t* Bt, int Mr, int Nc, int K,
                    const float* bias, int relu, const float* exBN,
                    const float* exBM, float* outF, half_t* outH,
                    const int* ts) {
        dim3 g(Nc / 16, (Mr + 255) / 256);
        size_t lds = (size_t)(K >> 4) * BROW_ + 32;
        gemm16<<<g, 128, lds, stream>>>(A, Bt, Mr, Nc, K, bias, relu, exBN,
                                        exBM, M_, N_, outF, outH, ts);
    };

    // --- pack weights (once per launch; deterministic) ---
    hipMemsetAsync(Wt_p1bn, 0, (size_t)512 * 1536 * 2, stream);
    pack(W_post1,    0,  256, 512, 512, Wt_p1bn, 1536,    0, 0); // target
    pack(W_post1,  512, 1024, 512, 512, Wt_p1bn, 1536,  256, 0); // payoff
    pack(W_post1, 1536,  240, 512, 512, Wt_p1bn, 1536, 1280, 0); // gt
    pack(W_post1,  256,  256, 512, 512, Wt_p1eg,  256,    0, 0); // ego slice
    pack(W_post2,    0,  512, 256, 256, Wt_p2,    512,    0, 0);
    pack(W_mu,       0,  256,  32,  32, Wt_mulv,  256,    0, 0);
    pack(W_lv,       0,  256,  32,  32, Wt_mulv,  256,    0, 32);
    pack(W_gen1,     0,  256, 512, 512, Wt_g1bn, 1280,    0, 0); // target
    pack(W_gen1,   512, 1024, 512, 512, Wt_g1bn, 1280,  256, 0); // payoff
    pack(W_gen1,   256,  256, 512, 512, Wt_g1eg,  256,    0, 0); // ego slice
    pack(W_gen1,  1536,   32, 512, 512, Wt_z,      32,    0, 0); // z slice
    pack(W_gen2,     0,  512, 256, 256, Wt_g2,    512,    0, 0);
    pack(W_gen3,     0,  256, 240, 240, Wt_g3,    256,    0, 0);
    pack(W_ego2,     0,  128, 256, 256, Wt_eg2,   128,    0, 0);

    // --- ego encoder: hbar = mean_t relu(x W1 + b1) ---
    ego1_kernel<<<RBM, 128, 0, stream>>>(ego, W_ego1, b_ego1, hbar16);

    // --- per-(b,n) features ---
    build_features<<<RBN, 256, 0, stream>>>(enc, gtf, nidx, nmask, tsp,
                                            Fpost, Fgen);

    // --- ego_ctx = hbar @ W_ego2 + b_ego2 (f16 out) ---
    gemm(hbar16, Wt_eg2, RBM, 256, 128, b_ego2, 0, nullptr, nullptr,
         nullptr, ego16, nullptr);

    // --- layer-1 partials (K-split of cond) ---
    gemm(Fpost, Wt_p1bn, RBN, 512, 1536, nullptr, 0, nullptr, nullptr,
         Pbn, nullptr, nullptr);
    gemm(Fgen, Wt_g1bn, RBN, 512, 1280, nullptr, 0, nullptr, nullptr,
         Gbn, nullptr, nullptr);
    gemm(ego16, Wt_p1eg, RBM, 512, 256, nullptr, 0, nullptr, nullptr,
         Pbm, nullptr, nullptr);
    gemm(ego16, Wt_g1eg, RBM, 512, 256, nullptr, 0, nullptr, nullptr,
         Gbm, nullptr, nullptr);

    // --- posterior net ---
    assemble_hp1<<<(RBMN * 512 + 255) / 256, 256, 0, stream>>>(
        Pbn, Pbm, b_post1, hp1);
    gemm(hp1, Wt_p2, RBMN, 256, 512, b_post2, 1, nullptr, nullptr,
         nullptr, hp2, nullptr);
    gemm(hp2, Wt_mulv, RBMN, 64, 256, nullptr, 0, nullptr, nullptr,
         mulv, nullptr, nullptr);
    z_kernel<<<(RBMN * LAT_ + 255) / 256, 256, 0, stream>>>(
        mulv, b_mu, b_lv, eps, z16);

    // --- generator net (z GEMM fused with broadcast-add of Gbn/Gbm) ---
    half_t* hg1 = hp1;
    half_t* hg2 = hp2;
    gemm(z16, Wt_z, RBMN, 512, 32, b_gen1, 1, Gbn, Gbm, nullptr, hg1, nullptr);
    gemm(hg1, Wt_g2, RBMN, 256, 512, b_gen2, 1, nullptr, nullptr,
         nullptr, hg2, nullptr);
    gemm(hg2, Wt_g3, RBMN, 240, 256, b_gen3, 0, nullptr, nullptr,
         (float*)d_out, nullptr, tsp);
}
